// CWVAE_21509196219238
// MI455X (gfx1250) — matrix operand order repaired and checked by reading, not compiled, not run on hardware
//
#include <hip/hip_runtime.h>
#include <math.h>

// ---------------------------------------------------------------------------
// CWVAE RSSM, CDNA5 (gfx1250).
//  * bf16 WMMA (v_wmma_f32_16x16x32_bf16) everywhere, f32 accumulate.
//  * Weights pre-transposed once to [N,K] bf16 (L2-resident) -> contiguous
//    16B B-fragment loads per lane.
//  * Dead prior head (w2/wpm/wps/np*) eliminated.
//  * concat(sample,ctx) fused into the w1 GEMM; top level runs K=128 only.
//  * gi (h@gru_wi) and gh (det@gru_wh) fused into one launch (independent).
// ---------------------------------------------------------------------------

typedef __bf16 bf16;
typedef __attribute__((ext_vector_type(16))) __bf16 v16bf;
typedef __attribute__((ext_vector_type(8)))  __bf16 v8bf;
typedef __attribute__((ext_vector_type(8)))  float  v8f;

union BF16x16 { v16bf v; v8bf h[2]; };

// One K-chunk: 4 stacked M-tiles against one B fragment.
__device__ __forceinline__ void wmma_chunk(
    const bf16* __restrict__ arow,   // A base for this lane: + col*strideA + hi*8
    int strideA, const bf16* __restrict__ bchunk, v8f acc[4])
{
  BF16x16 bfr;
  bfr.h[0] = *(const v8bf*)(bchunk);
  bfr.h[1] = *(const v8bf*)(bchunk + 16);
#pragma unroll
  for (int mt = 0; mt < 4; ++mt) {
    BF16x16 afr;
    const bf16* ap = arow + (size_t)(mt * 16) * strideA;
    afr.h[0] = *(const v8bf*)(ap);
    afr.h[1] = *(const v8bf*)(ap + 16);
    acc[mt] = __builtin_amdgcn_wmma_f32_16x16x32_bf16(
        false, afr.v, false, bfr.v, (short)0, acc[mt], false, false);
  }
}

__device__ __forceinline__ void gemm_epilogue(
    v8f acc[4], const float* __restrict__ bias, float* __restrict__ Cf,
    bf16* __restrict__ Cb, int N, int act, int n, int hi)
{
  const float bn = bias ? bias[n] : 0.f;
#pragma unroll
  for (int mt = 0; mt < 4; ++mt) {
#pragma unroll
    for (int r = 0; r < 8; ++r) {
      int m = mt * 16 + hi * 8 + r;            // f32 C/D layout: vgpr r -> M
      float v = acc[mt][r] + bn;
      if (act == 1) v = v > 0.f ? v : (expf(v) - 1.f);   // ELU
      size_t off = (size_t)m * N + n;
      if (Cf) Cf[off] = v;
      if (Cb) Cb[off] = (bf16)v;
    }
  }
}

// Generic GEMM body: C[64,N] = act(A[64,K] @ WT[N,K]^T + bias).
__device__ __forceinline__ void gemm_body(
    const bf16* __restrict__ A, const bf16* __restrict__ WT,
    const float* __restrict__ bias, float* __restrict__ Cf,
    bf16* __restrict__ Cb, int K, int N, int act, int blk)
{
  const int lane = threadIdx.x & 31;
  const int wave = threadIdx.x >> 5;
  const int col  = lane & 15;
  const int hi   = lane >> 4;
  const int n    = blk * 64 + wave * 16 + col;

  v8f acc[4] = {};
  const bf16* brow = WT + (size_t)n * K + hi * 8;
  const bf16* a0   = A + (size_t)col * K + hi * 8;
  for (int kc = 0; kc < K; kc += 32)
    wmma_chunk(a0 + kc, K, brow + kc, acc);
  gemm_epilogue(acc, bias, Cf, Cb, N, act, n, hi);
}

// Plain GEMM kernel (posterior path: q1, q2, wq).
__global__ __launch_bounds__(128) void k_gemm(
    const bf16* __restrict__ A, const bf16* __restrict__ WT,
    const float* __restrict__ bias, float* __restrict__ Cf,
    bf16* __restrict__ Cb, int K, int N, int act)
{
  gemm_body(A, WT, bias, Cf, Cb, K, N, act, blockIdx.x);
}

// w1 GEMM with fused concat(sample[64,128], ctx[64,1152]) A operand.
// Keff = 128 (top level, zero context) or 1280.
__global__ __launch_bounds__(128) void k_gemm_w1(
    const bf16* __restrict__ smp, const bf16* __restrict__ ctx,
    const bf16* __restrict__ WT /*[1024,1280]*/, const float* __restrict__ bias,
    bf16* __restrict__ Cb, int Keff)
{
  const int lane = threadIdx.x & 31;
  const int wave = threadIdx.x >> 5;
  const int col  = lane & 15;
  const int hi   = lane >> 4;
  const int n    = blockIdx.x * 64 + wave * 16 + col;

  v8f acc[4] = {};
  const bf16* brow = WT + (size_t)n * 1280 + hi * 8;
  // sample part: K rows of stride 128
  {
    const bf16* a0 = smp + (size_t)col * 128 + hi * 8;
#pragma unroll
    for (int kc = 0; kc < 128; kc += 32)
      wmma_chunk(a0 + kc, 128, brow + kc, acc);
  }
  // context part: K rows of stride 1152 (skipped entirely at top level)
  if (Keff > 128) {
    const bf16* a0 = ctx + (size_t)col * 1152 + hi * 8;
    for (int kc = 0; kc < 1152; kc += 32)
      wmma_chunk(a0 + kc, 1152, brow + 128 + kc, acc);
  }
  gemm_epilogue(acc, bias, nullptr, Cb, 1024, /*act=*/1, n, hi);
}

// Fused gi/gh launch: blocks 0..47 -> gi = h@wiT, blocks 48..95 -> gh = det@whT.
__global__ __launch_bounds__(128) void k_gemm_pair(
    const bf16* __restrict__ A0, const bf16* __restrict__ WT0,
    const float* __restrict__ b0, float* __restrict__ C0,
    const bf16* __restrict__ A1, const bf16* __restrict__ WT1,
    const float* __restrict__ b1, float* __restrict__ C1)
{
  const int sel = (blockIdx.x >= 48);
  const int blk = blockIdx.x - sel * 48;
  if (sel) gemm_body(A1, WT1, b1, C1, nullptr, 1024, 3072, 0, blk);
  else     gemm_body(A0, WT0, b0, C0, nullptr, 1024, 3072, 0, blk);
}

// Transpose f32 [K,N] -> bf16 [N,K] (one-time weight prep).
__global__ void k_transpose(const float* __restrict__ W, bf16* __restrict__ WT,
                            int K, int N)
{
  int idx = blockIdx.x * blockDim.x + threadIdx.x;
  if (idx >= K * N) return;
  int k = idx / N, n = idx % N;
  WT[(size_t)n * K + k] = (bf16)W[idx];
}

__global__ void k_pack_bq(const float* __restrict__ bqm,
                          const float* __restrict__ bqs, float* __restrict__ bq)
{
  int i = threadIdx.x;
  bq[i] = (i < 128) ? bqm[i] : bqs[i - 128];
}

// GRU combine + stage posterior GEMM input + emit det outputs.
__global__ void k_gru(const float* __restrict__ gi, const float* __restrict__ gh,
                      float* __restrict__ detf, bf16* __restrict__ detb,
                      bf16* __restrict__ Aq, bf16* __restrict__ out_t,
                      const float* __restrict__ x, int T, int t,
                      float* __restrict__ dout)
{
  int idx = blockIdx.x * blockDim.x + threadIdx.x;   // 64*1024
  int b = idx >> 10, d = idx & 1023;
  float gir = gi[b * 3072 + d];
  float giz = gi[b * 3072 + 1024 + d];
  float gin = gi[b * 3072 + 2048 + d];
  float ghr = gh[b * 3072 + d];
  float ghz = gh[b * 3072 + 1024 + d];
  float ghn = gh[b * 3072 + 2048 + d];
  float r  = 1.f / (1.f + expf(-(gir + ghr)));
  float z  = 1.f / (1.f + expf(-(giz + ghz)));
  float nn = tanhf(gin + r * ghn);
  float dn = (1.f - z) * nn + z * detf[idx];
  detf[idx] = dn;
  detb[idx] = (bf16)dn;
  Aq[b * 2048 + d]        = (bf16)dn;                                            // det_new
  Aq[b * 2048 + 1024 + d] = (bf16)x[(size_t)b * T * 1024 + (size_t)t * 1024 + d]; // obs
  out_t[b * 1152 + 128 + d] = (bf16)dn;
  if (dout) dout[(size_t)b * 36 * 1152 + (size_t)t * 1152 + 128 + d] = dn;
}

// Posterior head: q_sample = mean + (softplus(s)+1e-4) * noise.
__global__ void k_head(const float* __restrict__ qo, const float* __restrict__ eq,
                       int T, int t, bf16* __restrict__ sample,
                       bf16* __restrict__ out_t, float* __restrict__ dout)
{
  int idx = blockIdx.x * blockDim.x + threadIdx.x;   // 64*128
  int b = idx >> 7, s = idx & 127;
  float m  = qo[b * 256 + s];
  float sp = qo[b * 256 + 128 + s];
  float st = ((sp > 30.f) ? sp : log1pf(expf(sp))) + 1e-4f;
  float qs = m + st * eq[(size_t)b * T * 128 + (size_t)t * 128 + s];
  sample[idx] = (bf16)qs;
  out_t[b * 1152 + s] = (bf16)qs;
  if (dout) dout[(size_t)b * 36 * 1152 + (size_t)t * 1152 + s] = qs;
}

// ---------------------------------------------------------------------------
extern "C" void kernel_launch(void* const* d_in, const int* in_sizes, int n_in,
                              void* d_out, int out_size, void* d_ws, size_t ws_size,
                              hipStream_t stream)
{
  // setup_inputs() dict order: x0,np0,nq0, x1,np1,nq1, x2,np2,nq2, then params.
  const float* x[3]  = {(const float*)d_in[0], (const float*)d_in[3], (const float*)d_in[6]};
  const float* nq[3] = {(const float*)d_in[2], (const float*)d_in[5], (const float*)d_in[8]};
  const float* w1  = (const float*)d_in[9];
  const float* b1  = (const float*)d_in[10];
  const float* gwi = (const float*)d_in[11];
  const float* gwh = (const float*)d_in[12];
  const float* gbi = (const float*)d_in[13];
  const float* gbh = (const float*)d_in[14];
  // d_in[15..20]: w2,b2,wpm,bpm,wps,bps -> dead code, skipped.
  const float* q1  = (const float*)d_in[21];
  const float* qb1 = (const float*)d_in[22];
  const float* q2  = (const float*)d_in[23];
  const float* qb2 = (const float*)d_in[24];
  const float* wqm = (const float*)d_in[25];
  const float* bqm = (const float*)d_in[26];
  const float* wqs = (const float*)d_in[27];
  const float* bqs = (const float*)d_in[28];
  float* dout = (float*)d_out;

  size_t cur = 0;
  auto alloc = [&](size_t bytes) -> void* {
    cur = (cur + 255) & ~(size_t)255;
    void* p = (char*)d_ws + cur;
    cur += bytes;
    return p;
  };

  const int Ts[3] = {36, 6, 1};
  bf16 *w1T[3], *wiT[3], *whT[3], *q1T[3], *q2T[3], *wqT[3];
  float* bq[3];
  bf16* outbf[3];
  for (int l = 0; l < 3; ++l) {
    w1T[l]   = (bf16*)alloc((size_t)1280 * 1024 * 2);
    wiT[l]   = (bf16*)alloc((size_t)1024 * 3072 * 2);
    whT[l]   = (bf16*)alloc((size_t)1024 * 3072 * 2);
    q1T[l]   = (bf16*)alloc((size_t)2048 * 1024 * 2);
    q2T[l]   = (bf16*)alloc((size_t)1024 * 1024 * 2);
    wqT[l]   = (bf16*)alloc((size_t)256  * 1024 * 2);
    bq[l]    = (float*)alloc(256 * 4);
    outbf[l] = (bf16*)alloc((size_t)Ts[l] * 64 * 1152 * 2);
  }
  bf16*  hbf  = (bf16*)alloc((size_t)64 * 1024 * 2);
  float* gi   = (float*)alloc((size_t)64 * 3072 * 4);
  float* gh   = (float*)alloc((size_t)64 * 3072 * 4);
  bf16*  Aq   = (bf16*)alloc((size_t)64 * 2048 * 2);
  bf16*  hq1  = (bf16*)alloc((size_t)64 * 1024 * 2);
  bf16*  hq2  = (bf16*)alloc((size_t)64 * 1024 * 2);
  float* qo   = (float*)alloc((size_t)64 * 256 * 4);
  bf16*  smp  = (bf16*)alloc((size_t)64 * 128 * 2);
  bf16*  detb = (bf16*)alloc((size_t)64 * 1024 * 2);
  float* detf = (float*)alloc((size_t)64 * 1024 * 4);

  // --- One-time weight prep: f32 [K,N] -> bf16 [N,K] (~63MB, L2-resident). ---
  auto tr = [&](const float* W, bf16* WT, int K, int N) {
    int total = K * N;
    k_transpose<<<(total + 255) / 256, 256, 0, stream>>>(W, WT, K, N);
  };
  for (int l = 0; l < 3; ++l) {
    tr(w1  + (size_t)l * 1280 * 1024, w1T[l], 1280, 1024);
    tr(gwi + (size_t)l * 1024 * 3072, wiT[l], 1024, 3072);
    tr(gwh + (size_t)l * 1024 * 3072, whT[l], 1024, 3072);
    tr(q1  + (size_t)l * 2048 * 1024, q1T[l], 2048, 1024);
    tr(q2  + (size_t)l * 1024 * 1024, q2T[l], 1024, 1024);
    tr(wqm + (size_t)l * 1024 * 128,  wqT[l],                      1024, 128);
    tr(wqs + (size_t)l * 1024 * 128,  wqT[l] + (size_t)128 * 1024, 1024, 128);
    k_pack_bq<<<1, 256, 0, stream>>>(bqm + l * 128, bqs + l * 128, bq[l]);
  }

  // --- Top-down over levels; serial scan over time within each level. ---
  for (int l = 2; l >= 0; --l) {
    hipMemsetAsync(smp,  0, (size_t)64 * 128  * 2, stream);
    hipMemsetAsync(detb, 0, (size_t)64 * 1024 * 2, stream);
    hipMemsetAsync(detf, 0, (size_t)64 * 1024 * 4, stream);
    const float* b1l  = b1  + (size_t)l * 1024;
    const float* gbil = gbi + (size_t)l * 3072;
    const float* gbhl = gbh + (size_t)l * 3072;
    const float* qb1l = qb1 + (size_t)l * 1024;
    const float* qb2l = qb2 + (size_t)l * 1024;
    float* d0 = (l == 0) ? dout : nullptr;
    const int Keff = (l == 2) ? 128 : 1280;    // top level: context == 0

    for (int t = 0; t < Ts[l]; ++t) {
      const bf16* ctx = (l < 2) ? outbf[l + 1] + (size_t)(t / 6) * 64 * 1152
                                : (const bf16*)nullptr;
      bf16* ot = outbf[l] + (size_t)t * 64 * 1152;

      // prior trunk (posterior carry path only; dead prior head removed)
      k_gemm_w1<<<16, 128, 0, stream>>>(smp, ctx, w1T[l], b1l, hbf, Keff);
      k_gemm_pair<<<96, 128, 0, stream>>>(hbf,  wiT[l], gbil, gi,
                                          detb, whT[l], gbhl, gh);
      k_gru<<<256, 256, 0, stream>>>(gi, gh, detf, detb, Aq, ot, x[l], Ts[l], t, d0);

      // posterior
      k_gemm<<<16, 128, 0, stream>>>(Aq,  q1T[l], qb1l, nullptr, hq1, 2048, 1024, 1);
      k_gemm<<<16, 128, 0, stream>>>(hq1, q2T[l], qb2l, nullptr, hq2, 1024, 1024, 1);
      k_gemm<<<4,  128, 0, stream>>>(hq2, wqT[l], bq[l], qo, nullptr, 1024, 256, 0);
      k_head<<<32, 256, 0, stream>>>(qo, nq[l], Ts[l], t, smp, ot, d0);
    }
  }
}
